// CrossAttention_876173328968
// MI455X (gfx1250) — matrix operand order repaired and checked by reading, not compiled, not run on hardware
//
#include <hip/hip_runtime.h>

// ---------------------------------------------------------------------------
// CDNA5 (gfx1250) cross-attention, bf16 WMMA pipeline.
//   ln -> (Q,K,V) projections -> flash attention -> O projection
// Matrix math: v_wmma_f32_16x16x32_bf16 (f32 accumulate).
// Attention K/V chunks staged into LDS once per workgroup via
// global_load_async_to_lds_b128 (ASYNCcnt, double buffered).
// Workspace: ~771 MB of d_ws.
// ---------------------------------------------------------------------------

#define DEVFN static __device__ __forceinline__

typedef __attribute__((ext_vector_type(16))) __bf16        bf16x16;
typedef __attribute__((ext_vector_type(8)))  float         f32x8;
typedef __attribute__((ext_vector_type(4)))  unsigned int  u32x4;

// Problem constants (match reference).
constexpr int CB   = 8;       // batch
constexpr int CLQ  = 512;     // query length
constexpr int CLKV = 16384;   // kv length
constexpr int CLAT = 1024;    // latent channels
constexpr int CIN  = 768;     // input channels
constexpr int CQK  = 1024;    // qk channels
constexpr int CV   = 1024;    // v channels
constexpr int CH   = 8;       // heads
constexpr int CD   = 128;     // head dim
constexpr float CEPS = 1e-5f;

// ---------------------------------------------------------------------------
// WMMA helpers.  Layouts per CDNA5 ISA §7.12.2 (wave32):
//   A 16x32 bf16 : lane l -> row m=l&15; lanes 0-15 hold K {0..7,16..23},
//                  lanes 16-31 hold K {8..15,24..31}; packed 2/VGPR.
//   B 32x16 bf16 : lane l -> col n=l&15; lanes 0-15 hold K {0..7,16..23},
//                  lanes 16-31 hold K {8..15,24..31}; packed 2/VGPR.
//   C/D 16x16 f32: lane l -> n=l&15, VGPR v -> m = v + 8*(l>>4).
// ---------------------------------------------------------------------------

DEVFN f32x8 wmma_bf16(bf16x16 a, bf16x16 b, f32x8 c) {
  return __builtin_amdgcn_wmma_f32_16x16x32_bf16(
      /*neg_a=*/false, a, /*neg_b=*/false, b,
      /*c_mod=*/(short)0, c, /*reuse_a=*/false, /*reuse_b=*/false);
}

DEVFN bf16x16 load_pair_b128(const __bf16* p0, const __bf16* p1) {
  union { bf16x16 v; u32x4 q[2]; } u;
  u.q[0] = *(const u32x4*)p0;   // 16B -> global_load_b128 / ds_load_b128
  u.q[1] = *(const u32x4*)p1;
  return u.v;
}

// A-operand 16x32 tile from row-major [*, ld] bf16 at (row0, col0).
DEVFN bf16x16 load_a16x32(const __bf16* __restrict__ A, int ld, long row0, int col0) {
  const int l = threadIdx.x & 31;
  const __bf16* base = A + (row0 + (l & 15)) * (long)ld + col0 + ((l >> 4) * 8);
  return load_pair_b128(base, base + 16);
}

// B-operand 32x16 tile from row-major Bt [N, ld] at (n0, k0).
DEVFN bf16x16 load_b32x16(const __bf16* __restrict__ Bt, long ld, long n0, int k0) {
  const int l = threadIdx.x & 31;
  const __bf16* base = Bt + (n0 + (l & 15)) * ld + k0 + ((l >> 4) * 8);
  return load_pair_b128(base, base + 16);
}

// B-operand tile straight from an LDS-resident chunk (row-major, padded ld).
DEVFN bf16x16 lds_b32x16(const __bf16* buf, int ld, int n0, int k0) {
  const int l = threadIdx.x & 31;
  const __bf16* base = buf + (n0 + (l & 15)) * ld + k0 + ((l >> 4) * 8);
  return load_pair_b128(base, base + 16);
}

// Async global -> LDS 16B copy (per-lane), tracked by ASYNCcnt.
DEVFN void async_cp_b128(void* lds, const void* g) {
  asm volatile("global_load_async_to_lds_b128 %0, %1, off"
               :: "v"((unsigned)(size_t)lds), "v"(g)
               : "memory");
}

// 16-lane-half reductions (C layout rows live in a 16-lane half).
DEVFN float half16_max(float x) {
#pragma unroll
  for (int off = 1; off < 16; off <<= 1) x = fmaxf(x, __shfl_xor(x, off, 32));
  return x;
}
DEVFN float half16_sum(float x) {
#pragma unroll
  for (int off = 1; off < 16; off <<= 1) x += __shfl_xor(x, off, 32);
  return x;
}

// ---------------------------------------------------------------------------
// Kernel 1: LayerNorm fp32 row -> bf16 row.  One 256-thread block per row.
// ---------------------------------------------------------------------------
template <int CHN>
__global__ __launch_bounds__(256)
void k_layernorm_bf16(const float* __restrict__ x, const float* __restrict__ g,
                      const float* __restrict__ b, __bf16* __restrict__ y) {
  constexpr int PER = CHN / 256;
  const long row = blockIdx.x;
  const int  t   = threadIdx.x;
  __shared__ float s1[256], s2[256];

  float xv[PER];
  float acc = 0.f, acc2 = 0.f;
#pragma unroll
  for (int i = 0; i < PER; ++i) {
    float v = x[row * CHN + t + i * 256];
    xv[i] = v; acc += v; acc2 += v * v;
  }
  s1[t] = acc; s2[t] = acc2;
  __syncthreads();
#pragma unroll
  for (int s = 128; s > 0; s >>= 1) {
    if (t < s) { s1[t] += s1[t + s]; s2[t] += s2[t + s]; }
    __syncthreads();
  }
  const float mean = s1[0] * (1.0f / CHN);
  const float var  = s2[0] * (1.0f / CHN) - mean * mean;
  const float rstd = rsqrtf(var + CEPS);
#pragma unroll
  for (int i = 0; i < PER; ++i) {
    int c = t + i * 256;
    y[row * CHN + c] = (__bf16)((xv[i] - mean) * rstd * g[c] + b[c]);
  }
}

// ---------------------------------------------------------------------------
// Kernel 2: weight fp32 [K,N] -> bf16 transposed [N,K] (coalesced writes).
// ---------------------------------------------------------------------------
__global__ __launch_bounds__(256)
void k_weight_t_bf16(const float* __restrict__ W, __bf16* __restrict__ Wt,
                     int K, int N) {
  long idx = (long)blockIdx.x * 256 + threadIdx.x;
  if (idx < (long)K * N) {
    int n = (int)(idx / K), k = (int)(idx % K);
    Wt[idx] = (__bf16)W[(long)k * N + n];
  }
}

// ---------------------------------------------------------------------------
// Kernel 3: GEMM  C[M,N] = A[M,K](bf16) @ Wt[N,K](bf16)^T
// One wave computes a 32x64 tile: 2 A tiles, 4 B tiles, 8 WMMAs per K-step.
// Block (32,4): 4 waves stacked along M.
// mode: 0 = bf16 C, 1 = f32 C, 2 = bf16 coalesced scatter into V^T layout
//       [ (b*H+h)*128+d , Lkv ] (bounced through LDS -> 16B stores).
// ---------------------------------------------------------------------------
__global__ __launch_bounds__(128)
void k_gemm_bf16(const __bf16* __restrict__ A, int lda,
                 const __bf16* __restrict__ Bt, int ldb,
                 void* __restrict__ C, int ldc, int K, int mode) {
  constexpr int TSTR = 40;  // padded halves per staged column (bank-safe)
  __shared__ __attribute__((aligned(16))) __bf16 sT[4][64 * TSTR];

  const long m0 = ((long)blockIdx.y * 4 + threadIdx.y) * 32;
  const int  n0 = blockIdx.x * 64;
  const int  l = threadIdx.x & 31, half = l >> 4, ln = l & 15;

  f32x8 acc0[4] = {}, acc1[4] = {};
  for (int k0 = 0; k0 < K; k0 += 32) {
    bf16x16 a0 = load_a16x32(A, lda, m0, k0);
    bf16x16 a1 = load_a16x32(A, lda, m0 + 16, k0);
#pragma unroll
    for (int j = 0; j < 4; ++j) {
      bf16x16 bt = load_b32x16(Bt, ldb, n0 + j * 16, k0);
      acc0[j] = wmma_bf16(a0, bt, acc0[j]);
      acc1[j] = wmma_bf16(a1, bt, acc1[j]);
    }
  }

  __bf16* st = &sT[threadIdx.y][0];
#pragma unroll
  for (int i = 0; i < 2; ++i) {
    const f32x8* acc = i ? acc1 : acc0;
#pragma unroll
    for (int j = 0; j < 4; ++j) {
#pragma unroll
      for (int v = 0; v < 8; ++v) {
        int  ml = i * 16 + v + 8 * half;      // 0..31
        int  nl = j * 16 + ln;                // 0..63
        float val = acc[j][v];
        if (mode == 0) {
          ((__bf16*)C)[(m0 + ml) * (long)ldc + n0 + nl] = (__bf16)val;
        } else if (mode == 1) {
          ((float*)C)[(m0 + ml) * (long)ldc + n0 + nl] = val;
        } else {
          st[nl * TSTR + ml] = (__bf16)val;   // transpose via LDS
        }
      }
    }
  }
  if (mode == 2) {
    asm volatile("s_wait_dscnt 0" ::: "memory");  // per-wave LDS RAW fence
    const long bb   = m0 >> 14;                   // / LKV
    const long key0 = m0 & (CLKV - 1);
#pragma unroll
    for (int r = 0; r < 8; ++r) {
      int chunk = l + 32 * r;                     // 0..255
      int nl = chunk >> 2;                        // 0..63
      int mc = (chunk & 3) * 8;                   // 0,8,16,24
      u32x4 d = *(const u32x4*)&st[nl * TSTR + mc];
      int ng = n0 + nl;
      long vr = (bb * CH + (ng >> 7)) * CD + (ng & 127);
      *(u32x4*)((__bf16*)C + vr * (long)CLKV + key0 + mc) = d;
    }
  }
}

// ---------------------------------------------------------------------------
// Kernel 4: flash attention.  Block = 8 waves, all on the same (b,h); wave w
// owns q-tile qt = group*8+w.  Per 32-key chunk the block async-copies the
// K chunk (32x128) and V^T chunk (128x32) into LDS once (double buffered,
// ASYNCcnt), then each wave: 8 WMMA for S = Q K^T, online softmax in C
// layout, P bounced via per-wave LDS to A layout, 8 WMMA for O += P V.
// ---------------------------------------------------------------------------
__global__ __launch_bounds__(256)
void k_flash_attn(const __bf16* __restrict__ Q,   // [B*LQ , QK]
                  const __bf16* __restrict__ Km,  // [B*LKV, QK]
                  const __bf16* __restrict__ VT,  // [B*H*CD, LKV]
                  __bf16* __restrict__ Oatt) {    // [B*LQ , V]
  constexpr int NCHUNK = CLKV / 32;               // 512
  constexpr int KSTR = 136;  // 32x128 chunk, padded rows (272B: bank-safe)
  constexpr int VSTR = 40;   // 128x32 chunk, padded rows (80B: bank-safe)
  constexpr int PSTR = 40;   // 16x32 P tile per wave, padded

  __shared__ __attribute__((aligned(16))) __bf16 sK[2][32 * KSTR];
  __shared__ __attribute__((aligned(16))) __bf16 sV[2][128 * VSTR];
  __shared__ __attribute__((aligned(16))) __bf16 sP[8][16 * PSTR];

  const int w  = threadIdx.x >> 5;                // wave 0..7
  const int qg = blockIdx.x & 3;                  // q-tile group
  const int h  = (blockIdx.x >> 2) & 7;
  const int b  = blockIdx.x >> 5;
  const int qt = qg * 8 + w;

  const int l = threadIdx.x & 31, half = l >> 4, ln = l & 15;
  const float SCALE = 0.08838834764831845f;       // 1/sqrt(128)

  const long kbase = (long)b * CLKV;              // K row base
  const int  hoff  = h * CD;
  const long vbase = ((long)b * CH + h) * CD;     // V^T row base

  // Cooperative async stage of one 32-key chunk (4 x b128 per thread).
  auto issue = [&](int cc) {
    const int buf = cc & 1;
    const int k0  = cc * 32;
#pragma unroll
    for (int u = 0; u < 2; ++u) {
      int i = threadIdx.x + u * 256;
      int kr = i >> 4, kc = (i & 15) * 8;         // K: 32 rows x 128
      async_cp_b128(&sK[buf][kr * KSTR + kc],
                    Km + (kbase + k0 + kr) * (long)CQK + hoff + kc);
      int vr = i >> 2, vc = (i & 3) * 8;          // V^T: 128 rows x 32
      async_cp_b128(&sV[buf][vr * VSTR + vc],
                    VT + (vbase + vr) * (long)CLKV + k0 + vc);
    }
  };

  // Q tile: 16 x 128, four 16x32 A-operands (held in registers).
  bf16x16 qa[4];
#pragma unroll
  for (int kk = 0; kk < 4; ++kk)
    qa[kk] = load_a16x32(Q, CQK, (long)b * CLQ + qt * 16, hoff + kk * 32);

  f32x8 oc[8] = {};
  float mrun[8], lrun[8];
#pragma unroll
  for (int v = 0; v < 8; ++v) { mrun[v] = -INFINITY; lrun[v] = 0.f; }

  __bf16* pl = sP[w];
  issue(0);
  for (int c = 0; c < NCHUNK; ++c) {
    if (c + 1 < NCHUNK) {
      issue(c + 1);
      asm volatile("s_wait_asynccnt 4" ::: "memory"); // chunk c landed
    } else {
      asm volatile("s_wait_asynccnt 0" ::: "memory");
    }
    __syncthreads();                                   // chunk visible to all
    const __bf16* kbuf = sK[c & 1];
    const __bf16* vbuf = sV[c & 1];

    // --- S = Q K^T for the 32-key chunk (two 16-key column tiles) ---
    f32x8 s0 = {}, s1 = {};
#pragma unroll
    for (int kk = 0; kk < 4; ++kk) {
      bf16x16 kb0 = lds_b32x16(kbuf, KSTR, 0,  kk * 32);
      bf16x16 kb1 = lds_b32x16(kbuf, KSTR, 16, kk * 32);
      s0 = wmma_bf16(qa[kk], kb0, s0);
      s1 = wmma_bf16(qa[kk], kb1, s1);
    }

    // --- online softmax (C layout: row = v + 8*half, n = lane&15) ---
    float fac[8];
#pragma unroll
    for (int v = 0; v < 8; ++v) {
      float a0 = s0[v] * SCALE, a1 = s1[v] * SCALE;
      float mn = fmaxf(mrun[v], half16_max(fmaxf(a0, a1)));
      fac[v]   = __expf(mrun[v] - mn);
      mrun[v]  = mn;
      float p0 = __expf(a0 - mn), p1 = __expf(a1 - mn);
      s0[v] = p0; s1[v] = p1;
      lrun[v] = lrun[v] * fac[v] + half16_sum(p0 + p1);
    }
#pragma unroll
    for (int t = 0; t < 8; ++t)
#pragma unroll
      for (int v = 0; v < 8; ++v) oc[t][v] *= fac[v];

    // --- P: C layout -> per-wave LDS -> A layout ---
#pragma unroll
    for (int v = 0; v < 8; ++v) {
      int row = v + 8 * half;
      pl[row * PSTR + ln]      = (__bf16)s0[v];
      pl[row * PSTR + 16 + ln] = (__bf16)s1[v];
    }
    asm volatile("s_wait_dscnt 0" ::: "memory");  // same-wave LDS RAW fence
    const __bf16* pb = pl + ln * PSTR + half * 8;
    bf16x16 pa = load_pair_b128(pb, pb + 16);

    // --- O += P @ V ---
#pragma unroll
    for (int t = 0; t < 8; ++t) {
      bf16x16 vb = lds_b32x16(vbuf, VSTR, t * 16, 0);
      oc[t] = wmma_bf16(pa, vb, oc[t]);
    }
    __syncthreads();   // all waves done with this buffer before re-stage
  }

  // --- epilogue: normalize and store bf16 ---
#pragma unroll
  for (int v = 0; v < 8; ++v) lrun[v] = 1.0f / lrun[v];
#pragma unroll
  for (int t = 0; t < 8; ++t)
#pragma unroll
    for (int v = 0; v < 8; ++v) {
      long row = (long)b * CLQ + qt * 16 + v + 8 * half;
      int  col = hoff + t * 16 + ln;
      Oatt[row * CV + col] = (__bf16)(oc[t][v] * lrun[v]);
    }
}

// ---------------------------------------------------------------------------
// Host-side launcher.
// Inputs: latent_q, input_kv, W_Q, W_K, W_V, W_O, ln_lat_g, ln_lat_b,
//         ln_in_g, ln_in_b
// ---------------------------------------------------------------------------
extern "C" void kernel_launch(void* const* d_in, const int* in_sizes, int n_in,
                              void* d_out, int out_size, void* d_ws, size_t ws_size,
                              hipStream_t stream) {
  const float* latent = (const float*)d_in[0];
  const float* inpkv  = (const float*)d_in[1];
  const float* W_Q    = (const float*)d_in[2];
  const float* W_K    = (const float*)d_in[3];
  const float* W_V    = (const float*)d_in[4];
  const float* W_O    = (const float*)d_in[5];
  const float* g_lat  = (const float*)d_in[6];
  const float* b_lat  = (const float*)d_in[7];
  const float* g_in   = (const float*)d_in[8];
  const float* b_in   = (const float*)d_in[9];
  float* out = (float*)d_out;

  char* ws = (char*)d_ws;
  size_t off = 0;
  auto alloc = [&](size_t bytes) -> void* {
    void* p = ws + off;
    off += (bytes + 255) & ~(size_t)255;
    return p;
  };
  const size_t MQ = (size_t)CB * CLQ;    // 4096
  const size_t MK = (size_t)CB * CLKV;   // 131072
  __bf16* lnLat = (__bf16*)alloc(MQ * CLAT * 2);   //   8.4 MB
  __bf16* lnIn  = (__bf16*)alloc(MK * CIN  * 2);   // 201.3 MB
  __bf16* WtQ   = (__bf16*)alloc((size_t)CQK * CLAT * 2);
  __bf16* WtK   = (__bf16*)alloc((size_t)CQK * CIN  * 2);
  __bf16* WtV   = (__bf16*)alloc((size_t)CV  * CIN  * 2);
  __bf16* WtO   = (__bf16*)alloc((size_t)1024 * CV  * 2);
  __bf16* Qm    = (__bf16*)alloc(MQ * CQK * 2);    //   8.4 MB
  __bf16* Km    = (__bf16*)alloc(MK * CQK * 2);    // 268.4 MB
  __bf16* VTm   = (__bf16*)alloc(MK * CV  * 2);    // 268.4 MB
  __bf16* Oat   = (__bf16*)alloc(MQ * CV  * 2);    //   8.4 MB
  (void)ws_size; (void)in_sizes; (void)n_in; (void)out_size;

  // 1. layer norms -> bf16
  k_layernorm_bf16<CLAT><<<(int)MQ, 256, 0, stream>>>(latent, g_lat, b_lat, lnLat);
  k_layernorm_bf16<CIN ><<<(int)MK, 256, 0, stream>>>(inpkv,  g_in,  b_in,  lnIn);

  // 2. weights -> bf16 transposed [N,K]
  k_weight_t_bf16<<<(CLAT * CQK) / 256, 256, 0, stream>>>(W_Q, WtQ, CLAT, CQK);
  k_weight_t_bf16<<<(CIN  * CQK) / 256, 256, 0, stream>>>(W_K, WtK, CIN,  CQK);
  k_weight_t_bf16<<<(CIN  * CV ) / 256, 256, 0, stream>>>(W_V, WtV, CIN,  CV);
  k_weight_t_bf16<<<(CV * 1024) / 256, 256, 0, stream>>>(W_O, WtO, CV, 1024);

  // 3. projections (wave tile 32x64 -> grid (N/64, M/128), block (32,4))
  dim3 blk(32, 4, 1);
  k_gemm_bf16<<<dim3(CQK / 64, (unsigned)(MQ / 128)), blk, 0, stream>>>(
      lnLat, CLAT, WtQ, CLAT, Qm, CQK, CLAT, /*mode=*/0);
  k_gemm_bf16<<<dim3(CQK / 64, (unsigned)(MK / 128)), blk, 0, stream>>>(
      lnIn, CIN, WtK, CIN, Km, CQK, CIN, /*mode=*/0);
  k_gemm_bf16<<<dim3(CV / 64, (unsigned)(MK / 128)), blk, 0, stream>>>(
      lnIn, CIN, WtV, CIN, VTm, CV, CIN, /*mode=*/2);  // store V^T

  // 4. flash attention: 2048 q-tiles, 8 waves per block, shared K/V staging
  k_flash_attn<<<(CB * CH * (CLQ / 16)) / 8, 256, 0, stream>>>(Qm, Km, VTm, Oat);

  // 5. output projection, fp32 result
  k_gemm_bf16<<<dim3(1024 / 64, (unsigned)(MQ / 128)), blk, 0, stream>>>(
      Oat, CV, WtO, CV, out, 1024, CV, /*mode=*/1);
}
